// TransGNNBlock_35708358099445
// MI455X (gfx1250) — matrix operand order repaired and checked by reading, not compiled, not run on hardware
//
#include <hip/hip_runtime.h>
#include <hip/hip_bf16.h>
#include <math.h>

#define Nn   100000
#define Dd   64
#define Kk   20
#define Hh   4
#define Ee   3200000
#define DFFc 256

typedef __attribute__((ext_vector_type(16))) _Float16 v16h;
typedef __attribute__((ext_vector_type(8)))  _Float16 v8h;
typedef __attribute__((ext_vector_type(8)))  float    v8f;

// ---------------------------------------------------------------------------
// WMMA helpers (CDNA5 16x16x32 f16 -> f32), layouts per CDNA5 ISA
// ---------------------------------------------------------------------------
__device__ __forceinline__ v8f wmma16(v16h a, v16h b, v8f c) {
  // (neg_a, A, neg_b, B, c_mod, C, reuse_a, reuse_b)
  return __builtin_amdgcn_wmma_f32_16x16x32_f16(false, a, false, b, (short)0, c,
                                                false, false);
}

__device__ __forceinline__ v16h cat8(v8h lo, v8h hi) {
  return __builtin_shufflevector(lo, hi, 0, 1, 2, 3, 4, 5, 6, 7, 8, 9, 10, 11,
                                 12, 13, 14, 15);
}

__device__ __forceinline__ v8f splat8(float v) {
  v8f c;
#pragma unroll
  for (int r = 0; r < 8; ++r) c[r] = v;
  return c;
}

// A fragment (16x32, f16) from an LDS tile with row stride `ld` halves.
// lane&15 = M ; lane>>4 selects K-half ; element j -> K = 8*(lane>>4)+(j<8?j:j+8)
// => two contiguous 8-half runs: [kb, kb+8) and [kb+16, kb+24)  (b128 x2)
__device__ __forceinline__ v16h load_frag_A_lds(const _Float16* S, int ld, int kbase) {
  const int lane = threadIdx.x & 31;
  const int m  = lane & 15;
  const int kb = kbase + ((lane >> 4) << 3);
  const v8h lo = *(const v8h*)(S + m * ld + kb);
  const v8h hi = *(const v8h*)(S + m * ld + kb + 16);
  return cat8(lo, hi);
}

// B fragment (32x16) from a row-major *f16* weight W16[k][n], leading dim ldn.
// lane&15 = N ; element j -> K = 8*(lane>>4)+(j<8?j:j+8)  (strided scalar loads)
__device__ __forceinline__ v16h load_frag_B16(const _Float16* __restrict__ W16,
                                              int ldn, int kbase, int nbase) {
  const int lane = threadIdx.x & 31;
  const _Float16* p = W16 + (size_t)(kbase + ((lane >> 4) << 3)) * ldn + nbase + (lane & 15);
  v16h b;
#pragma unroll
  for (int j = 0; j < 16; ++j) b[j] = p[(size_t)(j < 8 ? j : j + 8) * ldn];
  return b;
}

// C/D fragment store: lane&15 = N ; VGPR r -> M = r + 8*(lane>>4)
__device__ __forceinline__ void store_frag_C_lds(float* S, int ld, int nbase, v8f c) {
  const int lane = threadIdx.x & 31;
  const int n  = nbase + (lane & 15);
  const int mb = (lane >> 4) << 3;
#pragma unroll
  for (int r = 0; r < 8; ++r) S[(mb + r) * ld + n] = c[r];
}

__device__ __forceinline__ void store_frag_C_g(float* __restrict__ O, int ld,
                                               long long rowbase, int nbase, v8f c) {
  const int lane = threadIdx.x & 31;
  const int n  = nbase + (lane & 15);
  const int mb = (lane >> 4) << 3;
#pragma unroll
  for (int r = 0; r < 8; ++r) O[(rowbase + mb + r) * (long long)ld + n] = c[r];
}

// Stage a 16xDd f32 global tile into an f16 LDS tile (b128 loads, packed cvt).
__device__ __forceinline__ void stage_tile_f16(const float* __restrict__ src,
                                               _Float16* dst) {
  const int lane = threadIdx.x & 31;
  const float4* s4 = (const float4*)src;
#pragma unroll
  for (int i = 0; i < 8; ++i) {
    const int e = i * 32 + lane;          // 256 float4 = 16*64 floats
    const float4 v = s4[e];
    _Float16* d = dst + e * 4;
    d[0] = (_Float16)v.x; d[1] = (_Float16)v.y;
    d[2] = (_Float16)v.z; d[3] = (_Float16)v.w;
  }
}

// LayerNorm over a 16x64 f32 LDS tile (row stride Dd). One wave: lane&15 = row,
// lane>>4 selects column half; cross-half combine via shfl_xor(16).
__device__ __forceinline__ void ln_tile(float* S, const float* __restrict__ g,
                                        const float* __restrict__ b) {
  const int lane = threadIdx.x & 31;
  const int row = lane & 15;
  const int c0  = (lane >> 4) * 32;
  float s = 0.f, s2 = 0.f;
#pragma unroll
  for (int c = 0; c < 32; ++c) {
    const float v = S[row * Dd + c0 + c];
    s += v; s2 += v * v;
  }
  s  += __shfl_xor(s, 16, 32);
  s2 += __shfl_xor(s2, 16, 32);
  const float mu  = s  * (1.f / Dd);
  const float var = s2 * (1.f / Dd) - mu * mu;
  const float inv = rsqrtf(var + 1e-5f);
#pragma unroll
  for (int c = 0; c < 32; ++c) {
    const int col = c0 + c;
    const float v = S[row * Dd + col];
    S[row * Dd + col] = (v - mu) * inv * g[col] + b[col];
  }
}

// ---------------------------------------------------------------------------
// Kernel 0: one-time fp32 -> f16 weight conversion (weights are tiny, then
// live in L2 as halves; B fragments are strided f16 scalar loads, no cvt)
// ---------------------------------------------------------------------------
__global__ void __launch_bounds__(256)
k_cvt_f16(const float* __restrict__ src, _Float16* __restrict__ dst, int n) {
  const int i = blockIdx.x * 256 + threadIdx.x;
  if (i < n) dst[i] = (_Float16)src[i];
}

// ---------------------------------------------------------------------------
// Kernel 1: q = x@Wq+bq ; xk = x@Wk+bk ; xv = x@Wv+bv   (gather commutes with
// the projection, so project the N rows once instead of the N*K gathered rows)
// ---------------------------------------------------------------------------
__global__ void __launch_bounds__(32)
k_proj_qkv(const float* __restrict__ x,
           const _Float16* __restrict__ Wq16, const float* __restrict__ bq,
           const _Float16* __restrict__ Wk16, const float* __restrict__ bk,
           const _Float16* __restrict__ Wv16, const float* __restrict__ bv,
           float* __restrict__ q, float* __restrict__ xk, float* __restrict__ xv) {
  __shared__ __align__(16) _Float16 A16[16 * Dd];
  const int lane = threadIdx.x;
  const long long rowbase = (long long)blockIdx.x * 16;

  stage_tile_f16(x + rowbase * Dd, A16);
  __syncthreads();

  const _Float16* Ws[3] = {Wq16, Wk16, Wv16};
  const float*    Bs[3] = {bq, bk, bv};
  float*          Os[3] = {q, xk, xv};
#pragma unroll
  for (int w = 0; w < 3; ++w) {
#pragma unroll
    for (int nt = 0; nt < 4; ++nt) {
      v8f c = splat8(Bs[w][nt * 16 + (lane & 15)]);
#pragma unroll
      for (int kc = 0; kc < 2; ++kc)
        c = wmma16(load_frag_A_lds(A16, Dd, kc * 32),
                   load_frag_B16(Ws[w], Dd, kc * 32, nt * 16), c);
      store_frag_C_g(Os[w], Dd, rowbase, nt * 16, c);
    }
  }
}

// ---------------------------------------------------------------------------
// Kernel 2: sampled attention. One wave32 per (node, head). Lanes 0..19 own one
// sample each (score), softmax via shuffles, then lanes 0..15 own one dim each.
// ---------------------------------------------------------------------------
__global__ void __launch_bounds__(256)
k_attn(const float* __restrict__ q, const float* __restrict__ xk,
       const float* __restrict__ xv, const int* __restrict__ samp,
       float* __restrict__ attn_o) {
  const int pair = blockIdx.x * 8 + (threadIdx.x >> 5);
  const int lane = threadIdx.x & 31;
  const int n = pair >> 2;
  const int h = pair & 3;

  float score = -3.0e38f;
  int idx = 0;
  if (lane < Kk) {
    idx = samp[(long long)n * Kk + lane];
    const float4* qr = (const float4*)(q  + (long long)n   * Dd + h * 16);
    const float4* kr = (const float4*)(xk + (long long)idx * Dd + h * 16);
    float s = 0.f;
#pragma unroll
    for (int d = 0; d < 4; ++d) {
      const float4 qv = qr[d], kv = kr[d];
      s += qv.x * kv.x + qv.y * kv.y + qv.z * kv.z + qv.w * kv.w;
    }
    score = s * 0.25f;  // 1/sqrt(DH=16)
  }
  float m = score;
#pragma unroll
  for (int off = 16; off > 0; off >>= 1) m = fmaxf(m, __shfl_xor(m, off, 32));
  const float e = (lane < Kk) ? __expf(score - m) : 0.f;
  float ssum = e;
#pragma unroll
  for (int off = 16; off > 0; off >>= 1) ssum += __shfl_xor(ssum, off, 32);
  const float a = e / ssum;

  float acc = 0.f;
  for (int k = 0; k < Kk; ++k) {
    const float ak = __shfl(a, k, 32);
    const int   ik = __shfl(idx, k, 32);
    if (lane < 16) acc += ak * xv[(long long)ik * Dd + h * 16 + lane];
  }
  if (lane < 16) attn_o[(long long)n * Dd + h * 16 + lane] = acc;
}

// ---------------------------------------------------------------------------
// Kernel 3: out@Wo+bo ; h1 = LN1(x+out) ; ffn = relu(h1@W1+bf1)@W2+bf2 ;
// h = LN2(h1+ffn). One wave per 16-row tile, all GEMMs via WMMA, streaming the
// 16x256 FFN intermediate through a 16x32 f16 LDS chunk.
// ---------------------------------------------------------------------------
__global__ void __launch_bounds__(32)
k_transformer_tail(const float* __restrict__ x, const float* __restrict__ attn_o,
                   const _Float16* __restrict__ Wo16, const float* __restrict__ bo,
                   const float* __restrict__ g1, const float* __restrict__ be1,
                   const _Float16* __restrict__ W116, const float* __restrict__ bf1,
                   const _Float16* __restrict__ W216, const float* __restrict__ bf2,
                   const float* __restrict__ g2, const float* __restrict__ be2,
                   float* __restrict__ hout) {
  __shared__ __align__(16) _Float16 A16[16 * Dd];
  __shared__ __align__(16) float    T32[16 * Dd];
  __shared__ __align__(16) float    H1 [16 * Dd];
  __shared__ __align__(16) _Float16 CH [16 * 32];
  const int lane = threadIdx.x;
  const long long base = (long long)blockIdx.x * 16 * Dd;

  stage_tile_f16(attn_o + base, A16);
  __syncthreads();

  // out @ Wo + bo
#pragma unroll
  for (int nt = 0; nt < 4; ++nt) {
    v8f c = splat8(bo[nt * 16 + (lane & 15)]);
#pragma unroll
    for (int kc = 0; kc < 2; ++kc)
      c = wmma16(load_frag_A_lds(A16, Dd, kc * 32),
                 load_frag_B16(Wo16, Dd, kc * 32, nt * 16), c);
    store_frag_C_lds(T32, Dd, nt * 16, c);
  }
  __syncthreads();
#pragma unroll
  for (int i = 0; i < 32; ++i) { const int e = i * 32 + lane; T32[e] += x[base + e]; }
  __syncthreads();
  ln_tile(T32, g1, be1);
  __syncthreads();
#pragma unroll
  for (int i = 0; i < 32; ++i) {
    const int e = i * 32 + lane;
    const float v = T32[e];
    H1[e] = v;
    A16[e] = (_Float16)v;
  }
  __syncthreads();

  // FFN: accumulate W2 output while streaming W1 output in 16x32 chunks
  v8f acc[4];
#pragma unroll
  for (int nt = 0; nt < 4; ++nt) acc[nt] = splat8(bf2[nt * 16 + (lane & 15)]);

  for (int kb = 0; kb < 8; ++kb) {
#pragma unroll
    for (int t = 0; t < 2; ++t) {
      const int ncol = kb * 32 + t * 16;
      v8f cc = splat8(bf1[ncol + (lane & 15)]);
#pragma unroll
      for (int kc = 0; kc < 2; ++kc)
        cc = wmma16(load_frag_A_lds(A16, Dd, kc * 32),
                    load_frag_B16(W116, DFFc, kc * 32, ncol), cc);
      const int nn = t * 16 + (lane & 15);
      const int mb = (lane >> 4) << 3;
#pragma unroll
      for (int r = 0; r < 8; ++r)
        CH[(mb + r) * 32 + nn] = (_Float16)fmaxf(cc[r], 0.f);  // ReLU
    }
    __syncthreads();
    const v16h aC = load_frag_A_lds(CH, 32, 0);
#pragma unroll
    for (int nt = 0; nt < 4; ++nt)
      acc[nt] = wmma16(aC, load_frag_B16(W216, Dd, kb * 32, nt * 16), acc[nt]);
    __syncthreads();
  }

#pragma unroll
  for (int nt = 0; nt < 4; ++nt) store_frag_C_lds(T32, Dd, nt * 16, acc[nt]);
  __syncthreads();
#pragma unroll
  for (int i = 0; i < 32; ++i) { const int e = i * 32 + lane; T32[e] += H1[e]; }
  __syncthreads();
  ln_tile(T32, g2, be2);
  __syncthreads();
#pragma unroll
  for (int i = 0; i < 32; ++i) { const int e = i * 32 + lane; hout[base + e] = T32[e]; }
}

// ---------------------------------------------------------------------------
// Kernel 4: edge scatter-add  h_nb[rows[e]] += edge_vals[e] * h[cols[e]]
// ---------------------------------------------------------------------------
__global__ void __launch_bounds__(256)
k_scatter(const int* __restrict__ rows, const int* __restrict__ cols,
          const float* __restrict__ ev, const float* __restrict__ h,
          float* __restrict__ hnb) {
  const long long t = (long long)blockIdx.x * 256 + threadIdx.x;
  const int e = (int)(t >> 6);
  const int d = (int)(t & 63);
  const float m = h[(long long)cols[e] * Dd + d] * ev[e];
  atomicAdd(&hnb[(long long)rows[e] * Dd + d], m);
}

// ---------------------------------------------------------------------------
// Kernel 5: h_new = relu(concat(h, h_nb) @ Wg + bg); out = LN(h_new + h)
// ---------------------------------------------------------------------------
__global__ void __launch_bounds__(32)
k_gnn_tail(const float* __restrict__ h, const float* __restrict__ hnb,
           const _Float16* __restrict__ Wg16, const float* __restrict__ bg,
           const float* __restrict__ gg, const float* __restrict__ bgn,
           float* __restrict__ out) {
  __shared__ __align__(16) _Float16 A16[16 * 128];
  __shared__ __align__(16) float    T32[16 * Dd];
  __shared__ __align__(16) float    HR [16 * Dd];
  const int lane = threadIdx.x;
  const long long base = (long long)blockIdx.x * 16 * Dd;

  // concat tile: cols 0..63 = h, 64..127 = h_nb (stride 128), plus f32 residual
  {
    const float4* h4 = (const float4*)(h + base);
    const float4* n4 = (const float4*)(hnb + base);
#pragma unroll
    for (int i = 0; i < 8; ++i) {
      const int e = i * 32 + lane;            // 256 float4 = 16*64 floats
      const int r = e >> 4, c4 = e & 15;      // row, float4-col
      const float4 hv = h4[e];
      const float4 nv = n4[e];
      float* hr = HR + e * 4;
      hr[0] = hv.x; hr[1] = hv.y; hr[2] = hv.z; hr[3] = hv.w;
      _Float16* da = A16 + r * 128 + c4 * 4;
      da[0] = (_Float16)hv.x; da[1] = (_Float16)hv.y;
      da[2] = (_Float16)hv.z; da[3] = (_Float16)hv.w;
      _Float16* db = A16 + r * 128 + Dd + c4 * 4;
      db[0] = (_Float16)nv.x; db[1] = (_Float16)nv.y;
      db[2] = (_Float16)nv.z; db[3] = (_Float16)nv.w;
    }
  }
  __syncthreads();

#pragma unroll
  for (int nt = 0; nt < 4; ++nt) {
    v8f c = splat8(bg[nt * 16 + (lane & 15)]);
#pragma unroll
    for (int kc = 0; kc < 4; ++kc)
      c = wmma16(load_frag_A_lds(A16, 128, kc * 32),
                 load_frag_B16(Wg16, Dd, kc * 32, nt * 16), c);
    const int n  = nt * 16 + (lane & 15);
    const int mb = (lane >> 4) << 3;
#pragma unroll
    for (int r = 0; r < 8; ++r) T32[(mb + r) * Dd + n] = fmaxf(c[r], 0.f);  // ReLU
  }
  __syncthreads();
#pragma unroll
  for (int i = 0; i < 32; ++i) { const int e = i * 32 + lane; T32[e] += HR[e]; }
  __syncthreads();
  ln_tile(T32, gg, bgn);
  __syncthreads();
#pragma unroll
  for (int i = 0; i < 32; ++i) { const int e = i * 32 + lane; out[base + e] = T32[e]; }
}

// ---------------------------------------------------------------------------
extern "C" void kernel_launch(void* const* d_in, const int* in_sizes, int n_in,
                              void* d_out, int out_size, void* d_ws, size_t ws_size,
                              hipStream_t stream) {
  const float* x    = (const float*)d_in[0];
  const int*   rows = (const int*)d_in[1];
  const int*   cols = (const int*)d_in[2];
  const float* ev   = (const float*)d_in[3];
  const int*   samp = (const int*)d_in[4];
  const float* Wq = (const float*)d_in[5];  const float* bq  = (const float*)d_in[6];
  const float* Wk = (const float*)d_in[7];  const float* bk  = (const float*)d_in[8];
  const float* Wv = (const float*)d_in[9];  const float* bv  = (const float*)d_in[10];
  const float* Wo = (const float*)d_in[11]; const float* bo  = (const float*)d_in[12];
  const float* g1 = (const float*)d_in[13]; const float* be1 = (const float*)d_in[14];
  const float* g2 = (const float*)d_in[15]; const float* be2 = (const float*)d_in[16];
  const float* W1 = (const float*)d_in[17]; const float* bf1 = (const float*)d_in[18];
  const float* W2 = (const float*)d_in[19]; const float* bf2 = (const float*)d_in[20];
  const float* Wg = (const float*)d_in[21]; const float* bg  = (const float*)d_in[22];
  const float* gg = (const float*)d_in[23]; const float* bgn = (const float*)d_in[24];

  // 4 x N*D f32 scratch buffers with reuse: B0=q->h, B1=xk->h_nb, B2=xv, B3=attn_out
  float* B0 = (float*)d_ws;
  float* B1 = B0 + (size_t)Nn * Dd;
  float* B2 = B1 + (size_t)Nn * Dd;
  float* B3 = B2 + (size_t)Nn * Dd;
  // f16 weight copies (converted once; ~115 KB total, L2-resident)
  _Float16* Wq16 = (_Float16*)(B3 + (size_t)Nn * Dd);
  _Float16* Wk16 = Wq16 + 64 * 64;
  _Float16* Wv16 = Wk16 + 64 * 64;
  _Float16* Wo16 = Wv16 + 64 * 64;
  _Float16* W116 = Wo16 + 64 * 64;
  _Float16* W216 = W116 + 64 * 256;
  _Float16* Wg16 = W216 + 256 * 64;

  k_cvt_f16<<<(64 * 64 + 255) / 256, 256, 0, stream>>>(Wq, Wq16, 64 * 64);
  k_cvt_f16<<<(64 * 64 + 255) / 256, 256, 0, stream>>>(Wk, Wk16, 64 * 64);
  k_cvt_f16<<<(64 * 64 + 255) / 256, 256, 0, stream>>>(Wv, Wv16, 64 * 64);
  k_cvt_f16<<<(64 * 64 + 255) / 256, 256, 0, stream>>>(Wo, Wo16, 64 * 64);
  k_cvt_f16<<<(64 * 256 + 255) / 256, 256, 0, stream>>>(W1, W116, 64 * 256);
  k_cvt_f16<<<(256 * 64 + 255) / 256, 256, 0, stream>>>(W2, W216, 256 * 64);
  k_cvt_f16<<<(128 * 64 + 255) / 256, 256, 0, stream>>>(Wg, Wg16, 128 * 64);

  k_proj_qkv<<<Nn / 16, 32, 0, stream>>>(x, Wq16, bq, Wk16, bk, Wv16, bv, B0, B1, B2);
  k_attn<<<(Nn * Hh) / 8, 256, 0, stream>>>(B0, B1, B2, samp, B3);
  k_transformer_tail<<<Nn / 16, 32, 0, stream>>>(x, B3, Wo16, bo, g1, be1, W116, bf1,
                                                 W216, bf2, g2, be2, B0);
  hipMemsetAsync(B1, 0, (size_t)Nn * Dd * sizeof(float), stream);
  k_scatter<<<(unsigned)((long long)Ee * Dd / 256), 256, 0, stream>>>(rows, cols, ev,
                                                                      B0, B1);
  k_gnn_tail<<<Nn / 16, 32, 0, stream>>>(B0, B1, Wg16, bg, gg, bgn, (float*)d_out);
  // second tuple element: attention_samples (int bits) appended after h_out
  hipMemcpyAsync((float*)d_out + (size_t)Nn * Dd, samp,
                 (size_t)Nn * Kk * sizeof(int), hipMemcpyDeviceToDevice, stream);
}
